// Attention_31817117729495
// MI455X (gfx1250) — compile-verified, matrix operand order checked by reading, not run on hardware
//
#include <hip/hip_runtime.h>
#include <hip/hip_bf16.h>

// ---------------- WMMA operand types (gfx1250, wave32) ----------------
typedef __attribute__((ext_vector_type(16))) __bf16 v16bf;
typedef __attribute__((ext_vector_type(2)))  __bf16 v2bf;
typedef __attribute__((ext_vector_type(8)))  float  v8f;
typedef __attribute__((ext_vector_type(2)))  float  v2f;
typedef __attribute__((ext_vector_type(8)))  int    v8i;
typedef __attribute__((ext_vector_type(4)))  int    v4i;

#define WAVE 32

// Problem sizes (fixed by the reference)
#define BATCH 8
#define SEQ   2048
#define DMODEL 1024          // D == H == 1024
#define MROWS (BATCH * SEQ)  // 16384

// f32 -> bf16 via native conversion (RNE); lowers to v_cvt*bf16 ops
__device__ __forceinline__ unsigned short f2bf(float f) {
    return __builtin_bit_cast(unsigned short, (__bf16)f);
}
// two f32 -> packed bf16x2 (one v_cvt_pk_bf16_f32)
__device__ __forceinline__ unsigned int pack_bf(float a, float b) {
    v2f f = {a, b};
    v2bf h = __builtin_convertvector(f, v2bf);
    return __builtin_bit_cast(unsigned int, h);
}

// 16-bit 16x32 A/B fragment (ISA 7.12.2). Per lane the 16 bf16 elements are
// two contiguous 16-byte chunks at ushort offsets 8*hi and 16+8*hi:
//   hi=0: K = 0..7  and 16..23      hi=1: K = 8..15 and 24..31
// => exactly two B128 loads per fragment. p must be 16-byte aligned and point
// at this lane's row start (row selected by lane&15).
__device__ __forceinline__ v16bf load_frag(const unsigned short* p, int hi) {
    const unsigned short* q0 = p + 8 * hi;
    v4i lo = *(const v4i*)(q0);
    v4i hh = *(const v4i*)(q0 + 16);
    v8i w  = __builtin_shufflevector(lo, hh, 0, 1, 2, 3, 4, 5, 6, 7);
    return __builtin_bit_cast(v16bf, w);
}

__device__ __forceinline__ v8f wmma_bf16(v16bf a, v16bf b, v8f c) {
    return __builtin_amdgcn_wmma_f32_16x16x32_bf16(
        /*neg_a=*/false, a, /*neg_b=*/false, b,
        /*c_mod=*/(short)0, c, /*reuse_a=*/false, /*reuse_b=*/false);
}

// =====================================================================
// Kernel 1: q/k/v = x @ W + b   (z = 0/1/2 selects the matrix)
//   q,k -> bf16 row-major [16384][1024]
//   v   -> bf16 transposed per batch: vt[b][h][n]  ([8][1024][2048])
// Workgroup: 128 thr (4 waves), 64x64 output tile, K-chunks of 32.
// =====================================================================
__global__ __launch_bounds__(128) void qkv_gemm_kernel(
    const float* __restrict__ x,
    const float* __restrict__ Wq, const float* __restrict__ bq,
    const float* __restrict__ Wk, const float* __restrict__ bk,
    const float* __restrict__ Wv, const float* __restrict__ bv,
    unsigned short* __restrict__ qo, unsigned short* __restrict__ ko,
    unsigned short* __restrict__ vt)
{
    // row stride 40 ushorts = 80 B (16B-aligned) so ds_load_b128 is legal
    __shared__ __align__(16) unsigned short As[64][40];   // [m][k]
    __shared__ __align__(16) unsigned short Bs[64][40];   // [n][k] (W transposed)

    const int z    = blockIdx.z;
    const float* W    = (z == 0) ? Wq : (z == 1) ? Wk : Wv;
    const float* bias = (z == 0) ? bq : (z == 1) ? bk : bv;

    const int tid  = threadIdx.x;
    const int w    = tid >> 5;        // wave 0..3
    const int lane = tid & (WAVE - 1);
    const int nlo  = lane & 15;
    const int hi   = lane >> 4;

    const int rowbase = blockIdx.x * 64;   // over M = 16384
    const int colbase = blockIdx.y * 64;   // over H = 1024

    v8f acc[4];
#pragma unroll
    for (int t = 0; t < 4; ++t)
#pragma unroll
        for (int r = 0; r < 8; ++r) acc[t][r] = 0.0f;

    for (int kc = 0; kc < DMODEL / 32; ++kc) {
        const int kk0 = kc * 32;
        // stage A: float4 (b128) reads of x, b64 LDS stores of 4 packed bf16
#pragma unroll
        for (int j = 0; j < 4; ++j) {
            const int e4 = tid + j * 128;          // 0..511 vec4s
            const int m = e4 >> 3, kv = (e4 & 7) * 4;
            float4 v = *(const float4*)&x[(rowbase + m) * DMODEL + kk0 + kv];
            *(uint2*)&As[m][kv] = make_uint2(pack_bf(v.x, v.y), pack_bf(v.z, v.w));
        }
        // stage B transposed: float4 reads of W row, scatter to 4 LDS rows
#pragma unroll
        for (int j = 0; j < 4; ++j) {
            const int e4 = tid + j * 128;
            const int n4 = (e4 & 15) * 4, ki = e4 >> 4;
            float4 v = *(const float4*)&W[(kk0 + ki) * DMODEL + colbase + n4];
            Bs[n4 + 0][ki] = f2bf(v.x);
            Bs[n4 + 1][ki] = f2bf(v.y);
            Bs[n4 + 2][ki] = f2bf(v.z);
            Bs[n4 + 3][ki] = f2bf(v.w);
        }
        // prefetch next K-chunk while this tile computes
        if (kc + 1 < DMODEL / 32) {
            __builtin_prefetch(&x[(rowbase + (tid >> 1)) * DMODEL + kk0 + 32], 0, 0);
            __builtin_prefetch(&W[(kk0 + 32 + (tid >> 2)) * DMODEL + colbase], 0, 0);
        }
        __syncthreads();

        v16bf a = load_frag(&As[w * 16 + nlo][0], hi);
#pragma unroll
        for (int t = 0; t < 4; ++t) {
            v16bf b = load_frag(&Bs[t * 16 + nlo][0], hi);
            acc[t] = wmma_bf16(a, b, acc[t]);
        }
        __syncthreads();
    }

    // epilogue: bias + bf16 store
#pragma unroll
    for (int t = 0; t < 4; ++t) {
        const int col = colbase + t * 16 + nlo;
        const float bv_ = bias[col];
#pragma unroll
        for (int r = 0; r < 8; ++r) {
            const int mrow = rowbase + w * 16 + r + 8 * hi;
            const unsigned short h = f2bf(acc[t][r] + bv_);
            if (z == 0)      qo[mrow * DMODEL + col] = h;
            else if (z == 1) ko[mrow * DMODEL + col] = h;
            else {
                const int bb = mrow >> 11, nn = mrow & (SEQ - 1);
                vt[((bb << 10) + col) * SEQ + nn] = h;   // [b][h][n]
            }
        }
    }
}

// score tile: C[16x16] = q_rows(16) . k_keys(16)^T over full D=1024
__device__ __forceinline__ v8f score_tile(
    const unsigned short* __restrict__ q, const unsigned short* __restrict__ k,
    int b, int qr0, int kb, int nlo, int hi)
{
    v8f c;
#pragma unroll
    for (int r = 0; r < 8; ++r) c[r] = 0.0f;
    const unsigned short* ap0 = q + (b * SEQ + qr0 + nlo)     * DMODEL;
    const unsigned short* bp0 = k + (b * SEQ + kb * 16 + nlo) * DMODEL;
#pragma unroll 4
    for (int dc = 0; dc < DMODEL / 32; ++dc) {
        v16bf a  = load_frag(ap0 + dc * 32, hi);
        v16bf bf = load_frag(bp0 + dc * 32, hi);
        c = wmma_bf16(a, bf, c);
    }
    return c;
}

// =====================================================================
// Kernel 2: P = softmax(mask(q k^T / sqrt(H)))  -> bf16 [B][N][N]
// 2-pass online softmax; 4 waves x 16 query rows per 64-row block.
// =====================================================================
__global__ __launch_bounds__(128) void attn_probs_kernel(
    const unsigned short* __restrict__ q, const unsigned short* __restrict__ k,
    unsigned short* __restrict__ P)
{
    const int tid  = threadIdx.x;
    const int w    = tid >> 5;
    const int lane = tid & (WAVE - 1);
    const int nlo  = lane & 15;
    const int hi   = lane >> 4;

    const int qbase = blockIdx.x * 64;
    const int b     = blockIdx.y;
    const int qr0   = qbase + w * 16;
    const int nkb   = (qbase >> 4) + 4;          // key tiles up to block diagonal
    const float scale = 0.03125f;                // 1/sqrt(1024)

    float m8[8], l8[8];
#pragma unroll
    for (int r = 0; r < 8; ++r) { m8[r] = -1e30f; l8[r] = 0.0f; }

    // ---- pass 1: row max / row sum-exp ----
    for (int kb = 0; kb < nkb; ++kb) {
        v8f c = score_tile(q, k, b, qr0, kb, nlo, hi);
#pragma unroll
        for (int r = 0; r < 8; ++r) {
            const int qrow = qr0 + r + 8 * hi;
            const int kcol = kb * 16 + nlo;
            float s = (kcol <= qrow) ? c[r] * scale : -1e30f;
            float tm = s;
            tm = fmaxf(tm, __shfl_xor(tm, 1));
            tm = fmaxf(tm, __shfl_xor(tm, 2));
            tm = fmaxf(tm, __shfl_xor(tm, 4));
            tm = fmaxf(tm, __shfl_xor(tm, 8));
            const float nm = fmaxf(m8[r], tm);
            float e = __expf(s - nm);
            e += __shfl_xor(e, 1);
            e += __shfl_xor(e, 2);
            e += __shfl_xor(e, 4);
            e += __shfl_xor(e, 8);
            l8[r] = l8[r] * __expf(m8[r] - nm) + e;
            m8[r] = nm;
        }
    }
    float rl8[8];
#pragma unroll
    for (int r = 0; r < 8; ++r) rl8[r] = 1.0f / l8[r];

    // ---- pass 2: recompute, normalize, emit bf16 probs ----
    for (int kb = 0; kb < nkb; ++kb) {
        v8f c = score_tile(q, k, b, qr0, kb, nlo, hi);
#pragma unroll
        for (int r = 0; r < 8; ++r) {
            const int qrow = qr0 + r + 8 * hi;
            const int kcol = kb * 16 + nlo;
            float p = (kcol <= qrow) ? __expf(c[r] * scale - m8[r]) * rl8[r]
                                     : 0.0f;
            P[(size_t)(b * SEQ + qrow) * SEQ + kcol] = f2bf(p);
        }
    }
}

// =====================================================================
// Kernel 3: out = P @ V   (per batch), K-loop bounded by causal diagonal.
//   A = P bf16 [N][N], B = vt bf16 [H][N] (key-pairs contiguous)
// =====================================================================
__global__ __launch_bounds__(128) void pv_gemm_kernel(
    const unsigned short* __restrict__ P, const unsigned short* __restrict__ vt,
    float* __restrict__ out)
{
    const int tid  = threadIdx.x;
    const int w    = tid >> 5;
    const int lane = tid & (WAVE - 1);
    const int nlo  = lane & 15;
    const int hi   = lane >> 4;

    const int htile = blockIdx.x;          // 0..15
    const int qbase = blockIdx.y * 64;     // 0..31 tiles
    const int b     = blockIdx.z;
    const int qr0   = qbase + w * 16;
    const int nkc   = (qbase >> 5) + 2;    // 32-key chunks up to diagonal

    v8f acc[4];
#pragma unroll
    for (int t = 0; t < 4; ++t)
#pragma unroll
        for (int r = 0; r < 8; ++r) acc[t][r] = 0.0f;

    const unsigned short* ap0 = P + (size_t)(b * SEQ + qr0 + nlo) * SEQ;
#pragma unroll 2
    for (int kc = 0; kc < nkc; ++kc) {
        if (kc + 1 < nkc) __builtin_prefetch(ap0 + (kc + 1) * 32, 0, 0);
        v16bf a = load_frag(ap0 + kc * 32, hi);
#pragma unroll
        for (int t = 0; t < 4; ++t) {
            const int hcol = htile * 64 + t * 16 + nlo;
            const unsigned short* bp =
                vt + (size_t)(b * DMODEL + hcol) * SEQ + kc * 32;
            v16bf bf = load_frag(bp, hi);
            acc[t] = wmma_bf16(a, bf, acc[t]);
        }
    }

#pragma unroll
    for (int t = 0; t < 4; ++t) {
        const int hcol = htile * 64 + t * 16 + nlo;
#pragma unroll
        for (int r = 0; r < 8; ++r) {
            const int qrow = qr0 + r + 8 * hi;
            out[(size_t)(b * SEQ + qrow) * DMODEL + hcol] = acc[t][r];
        }
    }
}

// =====================================================================
extern "C" void kernel_launch(void* const* d_in, const int* in_sizes, int n_in,
                              void* d_out, int out_size, void* d_ws, size_t ws_size,
                              hipStream_t stream) {
    const float* x  = (const float*)d_in[0];
    const float* Wq = (const float*)d_in[1];
    const float* bq = (const float*)d_in[2];
    const float* Wk = (const float*)d_in[3];
    const float* bk = (const float*)d_in[4];
    const float* Wv = (const float*)d_in[5];
    const float* bv = (const float*)d_in[6];
    float* out = (float*)d_out;

    // workspace layout (bf16 halves): q | k | vt | P   (~168 MB total)
    unsigned short* qb = (unsigned short*)d_ws;
    unsigned short* kb = qb + (size_t)MROWS * DMODEL;
    unsigned short* vt = kb + (size_t)MROWS * DMODEL;          // [B][H][N]
    unsigned short* P  = vt + (size_t)MROWS * DMODEL;          // [B][N][N]

    dim3 blk(128);
    dim3 g1(MROWS / 64, DMODEL / 64, 3);
    qkv_gemm_kernel<<<g1, blk, 0, stream>>>(x, Wq, bq, Wk, bk, Wv, bv,
                                            qb, kb, vt);
    dim3 g2(SEQ / 64, BATCH);
    attn_probs_kernel<<<g2, blk, 0, stream>>>(qb, kb, P);

    dim3 g3(DMODEL / 64, SEQ / 64, BATCH);
    pv_gemm_kernel<<<g3, blk, 0, stream>>>(P, vt, out);
}
